// PointOrientedAggregation_9663676416050
// MI455X (gfx1250) — compile-verified, matrix operand order checked by reading, not run on hardware
//
#include <hip/hip_runtime.h>

// ---------------------------------------------------------------------------
// PointOrientedAggregation for MI455X (gfx1250, wave32, WMMA)
// B=2, N=4096, H=256, K=24, IN_DIM=515
//
// feat@W1 factorizes into per-point terms:
//   x1(i,j) = leaky( p_ij@W1[0:3] + (h_i@W1[3:259]) + (h_j@W1[259:515]) + b1 )
// so we precompute U = h @ [W1_hi | W1_hj]  (S = U[:, :256], T = U[:, 256:])
// once for all 8192 points (~2 GF), cutting total FLOPs roughly in half
// (~52 GF bf16 WMMA for layers 2/3, ~17 GF f32 WMMA for kNN distances).
// HBM traffic ~40 MB (~2 us @ 23.3 TB/s) -> compute-bound on the WMMA pipes.
// All LDS tile staging is double-buffered (1 barrier per K-step) and the
// data-dependent neighbor rows are prefetched (global_prefetch_b8).
// ---------------------------------------------------------------------------

typedef __bf16 bf16_t;
typedef __attribute__((ext_vector_type(16))) __bf16 v16bf;
typedef __attribute__((ext_vector_type(8)))  __bf16 v8bf;
typedef __attribute__((ext_vector_type(8)))  float  v8f;
typedef __attribute__((ext_vector_type(4)))  float  v4f;
typedef __attribute__((ext_vector_type(2)))  float  v2f;

union BF16x16 { v16bf v; v8bf h[2]; };

#define WMMA_BF16(a, b, c) \
  __builtin_amdgcn_wmma_f32_16x16x32_bf16(false, (a), false, (b), (short)0, (c), false, false)
#define WMMA_F32(a, b, c) \
  __builtin_amdgcn_wmma_f32_16x16x4_f32(false, (a), false, (b), (short)0, (c), false, false)

#define NB 2
#define NPTS 4096
#define HD 256
#define KNN 24
#define NROWS (NB * NPTS)   // 8192

// ---------------------------------------------------------------------------
// Kernel 1: h (f32) -> hbf (bf16) + per-row sum of squares
// ---------------------------------------------------------------------------
__global__ __launch_bounds__(256) void pack_h_kernel(
    const float* __restrict__ h, bf16_t* __restrict__ hbf, float* __restrict__ sq) {
  __shared__ float red[256];
  const int row = blockIdx.x;
  const int t = threadIdx.x;
  float v = h[(size_t)row * HD + t];
  hbf[(size_t)row * HD + t] = (bf16_t)v;
  red[t] = v * v;
  __syncthreads();
  for (int s = 128; s > 0; s >>= 1) {
    if (t < s) red[t] += red[t + s];
    __syncthreads();
  }
  if (t == 0) sq[row] = red[0];
}

// ---------------------------------------------------------------------------
// Kernel 2: transposed bf16 weights:
//   Bt1[512][256]: rows 0..255 = W1[3+k][n], rows 256..511 = W1[259+k][n]
//   Bt2[n][k] = W2[k][n], Bt3[n][k] = W3[k][n]
// ---------------------------------------------------------------------------
__global__ __launch_bounds__(256) void pack_w_kernel(
    const float* __restrict__ W1, const float* __restrict__ W2, const float* __restrict__ W3,
    bf16_t* __restrict__ Bt1, bf16_t* __restrict__ Bt2, bf16_t* __restrict__ Bt3) {
  const int r = blockIdx.x;
  const int k = threadIdx.x;
  if (r < 512) {
    const int srcRow = (r < 256) ? (3 + k) : (259 + k);
    Bt1[(size_t)r * 256 + k] = (bf16_t)W1[(size_t)srcRow * 256 + (r & 255)];
  } else if (r < 768) {
    const int n = r - 512;
    Bt2[(size_t)n * 256 + k] = (bf16_t)W2[(size_t)k * 256 + n];
  } else {
    const int n = r - 768;
    Bt3[(size_t)n * 256 + k] = (bf16_t)W3[(size_t)k * 256 + n];
  }
}

// ---------------------------------------------------------------------------
// Kernel 3: kNN in hidden space via f32 WMMA dot products.
// WG = 128 thr (4 waves) owns 64 query rows; wave w owns 16 rows.
// Candidate tiles double-buffered: one barrier per 32-candidate tile.
// Ordering key: d' = |h_j|^2 - 2 h_i.h_j  (|h_i|^2 constant per row).
// dynamic LDS 154112 B
// ---------------------------------------------------------------------------
__global__ __launch_bounds__(128) void knn_kernel(
    const float* __restrict__ h, const float* __restrict__ sq, int* __restrict__ knn_idx) {
  extern __shared__ char smem[];
  float* hq   = (float*)smem;            // 64 x 260
  float* hc   = hq + 64 * 260;           // 2 x 32 x 260 (double buffer)
  float* sqc  = hc + 2 * 32 * 260;       // 2 x 32
  float* dist = sqc + 64;                // 4 waves x 16 x 33
  float* topd = dist + 4 * 16 * 33;      // 64 x 24
  int*   topi = (int*)(topd + 64 * KNN); // 64 x 24

  const int b = blockIdx.x >> 6;
  const int rowbase = (blockIdx.x & 63) * 64;
  const float* hb = h + (size_t)b * NPTS * HD;
  const int t = threadIdx.x;
  const int wave = t >> 5, lane = t & 31;
  const int lm = lane & 15, hi = lane >> 4;
  const int myrow = wave * 16 + lm;

  // stage 64 query rows (f32, exact), padded stride 260 (bank-conflict-free)
  {
    const int row = t >> 1;
    const int off = (t & 1) * 128;
    const float* src = hb + (size_t)(rowbase + row) * HD + off;
    float* dst = hq + row * 260 + off;
#pragma unroll
    for (int i = 0; i < 128; i += 4) *(v4f*)(dst + i) = *(const v4f*)(src + i);
  }
  for (int i = t; i < 64 * KNN; i += 128) { topd[i] = 3.0e38f; topi[i] = 0; }

  // candidate tile staging (double-buffered)
  auto stage_c = [&](int jt, int buf) {
    const int row = t >> 2;
    const int off = (t & 3) * 64;
    const float* src = hb + (size_t)(jt * 32 + row) * HD + off;
    float* dst = hc + buf * (32 * 260) + row * 260 + off;
#pragma unroll
    for (int i = 0; i < 64; i += 4) *(v4f*)(dst + i) = *(const v4f*)(src + i);
    if (t < 32) sqc[buf * 32 + t] = sq[b * NPTS + jt * 32 + t];
  };

  stage_c(0, 0);
  __syncthreads();

  float worst = 3.0e38f;
  int   wslot = 23;

  for (int jt = 0; jt < NPTS / 32; ++jt) {
    if (jt + 1 < NPTS / 32) stage_c(jt + 1, (jt + 1) & 1);

    const float* hcb = hc + (jt & 1) * (32 * 260);
    const float* sqb = sqc + (jt & 1) * 32;
    const float* aq = hq + (wave * 16 + lm) * 260;
    float* dw = dist + wave * (16 * 33);
#pragma unroll
    for (int nt = 0; nt < 2; ++nt) {
      v8f acc = {};
      const float* bc = hcb + (nt * 16 + lm) * 260;
#pragma unroll 8
      for (int s = 0; s < 64; ++s) {
        const int k0 = 4 * s + 2 * hi;
        v2f a  = *(const v2f*)(aq + k0);
        v2f bb = *(const v2f*)(bc + k0);
        acc = WMMA_F32(a, bb, acc);
      }
      const float sqn = sqb[nt * 16 + lm];
#pragma unroll
      for (int r = 0; r < 8; ++r)
        dw[(r + 8 * hi) * 33 + nt * 16 + lm] = sqn - 2.0f * acc[r];
    }

    // per-wave dist region: wave-internal LDS ordering suffices (no barrier)
    if (lane < 16) {
      const float* drow = dist + wave * (16 * 33) + lane * 33;
      float* td = topd + myrow * KNN;
      int*   ti = topi + myrow * KNN;
      for (int c = 0; c < 32; ++c) {
        const float d = drow[c];
        if (d < worst) {
          td[wslot] = d;
          ti[wslot] = jt * 32 + c;
          float wv = -3.0e38f; int ws = 0;
#pragma unroll
          for (int s2 = 0; s2 < KNN; ++s2) {
            float v = td[s2];
            if (v >= wv) { wv = v; ws = s2; }
          }
          worst = wv; wslot = ws;
        }
      }
    }
    __syncthreads();
  }
  for (int i = t; i < 64 * KNN; i += 128) {
    const int row = i / KNN, s = i % KNN;
    knn_idx[((size_t)b * NPTS + rowbase + row) * KNN + s] = topi[row * KNN + s];
  }
}

// ---------------------------------------------------------------------------
// Kernel 4: U[8192][512] = hbf[8192][256] @ Bt1^T  (bf16 WMMA, f32 accum)
// 128x128 tiles, Kc=32, double-buffered LDS staging (1 barrier per K-step).
// ---------------------------------------------------------------------------
__global__ __launch_bounds__(256) void gemm_u_kernel(
    const bf16_t* __restrict__ A, const bf16_t* __restrict__ Bt, float* __restrict__ U) {
  __shared__ bf16_t At[2][128 * 40];
  __shared__ bf16_t Btt[2][128 * 40];
  const int t = threadIdx.x;
  const int wave = t >> 5, lane = t & 31, lm = lane & 15, hi = lane >> 4;
  const int mbase = (blockIdx.x >> 2) * 128;
  const int nbase = (blockIdx.x & 3) * 128;
  v8f acc[8] = {};

  auto stage = [&](int kt, int buf) {
#pragma unroll
    for (int i = 0; i < 2; ++i) {
      const int id = t + i * 256;
      const int row = id >> 2, off = (id & 3) * 8;
      *(v8bf*)(&At[buf][row * 40 + off]) =
          *(const v8bf*)(A + (size_t)(mbase + row) * 256 + kt * 32 + off);
      *(v8bf*)(&Btt[buf][row * 40 + off]) =
          *(const v8bf*)(Bt + (size_t)(nbase + row) * 256 + kt * 32 + off);
    }
  };

  stage(0, 0);
  __syncthreads();
  for (int kt = 0; kt < 8; ++kt) {
    if (kt < 7) stage(kt + 1, (kt + 1) & 1);
    const bf16_t* Ac = At[kt & 1];
    const bf16_t* Bc = Btt[kt & 1];
    const int kb = hi * 8;
    BF16x16 af;
    const bf16_t* ap = Ac + (wave * 16 + lm) * 40;
    af.h[0] = *(const v8bf*)(ap + kb);
    af.h[1] = *(const v8bf*)(ap + 16 + kb);
#pragma unroll
    for (int nt = 0; nt < 8; ++nt) {
      BF16x16 bfv;
      const bf16_t* bp = Bc + (nt * 16 + lm) * 40;
      bfv.h[0] = *(const v8bf*)(bp + kb);
      bfv.h[1] = *(const v8bf*)(bp + 16 + kb);
      acc[nt] = WMMA_BF16(af.v, bfv.v, acc[nt]);
    }
    __syncthreads();
  }
#pragma unroll
  for (int nt = 0; nt < 8; ++nt)
#pragma unroll
    for (int r = 0; r < 8; ++r)
      U[(size_t)(mbase + wave * 16 + r + 8 * hi) * 512 + nbase + nt * 16 + lm] = acc[nt][r];
}

// ---------------------------------------------------------------------------
// Fused-kernel helper: one N-chunk (64 cols) of acc += Xin @ Wt^T.
// Xin: LDS bf16 stride 264.  Wt: global bf16 [256][256] (n-major).
// Btile double-buffered: one barrier per K-step.
// ---------------------------------------------------------------------------
template <int NMI>
__device__ __forceinline__ void gemm_chunk(
    const bf16_t* __restrict__ Xin, const bf16_t* __restrict__ Wt,
    bf16_t* __restrict__ Btile,   // 2 x (64*40)
    int nchunk, int mt0, int ntl, int lm, int hi, int t, v8f acc[NMI]) {
  const int row = t >> 2, off = (t & 3) * 8;
  const bf16_t* wrow = Wt + (size_t)(nchunk + row) * 256 + off;
  *(v8bf*)(Btile + row * 40 + off) = *(const v8bf*)(wrow);
  __syncthreads();
  for (int ks = 0; ks < 8; ++ks) {
    const int k0 = ks * 32;
    if (ks < 7) {
      *(v8bf*)(Btile + ((ks + 1) & 1) * (64 * 40) + row * 40 + off) =
          *(const v8bf*)(wrow + k0 + 32);
    }
    const bf16_t* cur = Btile + (ks & 1) * (64 * 40);
    const int kb = hi * 8;
    BF16x16 bfr;
    const bf16_t* bp = cur + (ntl * 16 + lm) * 40;
    bfr.h[0] = *(const v8bf*)(bp + kb);
    bfr.h[1] = *(const v8bf*)(bp + 16 + kb);
#pragma unroll
    for (int mi = 0; mi < NMI; ++mi) {
      const bf16_t* ap = Xin + ((mt0 + mi) * 16 + lm) * 264 + k0;
      BF16x16 afr;
      afr.h[0] = *(const v8bf*)(ap + kb);
      afr.h[1] = *(const v8bf*)(ap + 16 + kb);
      acc[mi] = WMMA_BF16(afr.v, bfr.v, acc[mi]);
    }
    __syncthreads();
  }
}

// ---------------------------------------------------------------------------
// Kernel 5: fused layer-1 assembly + layer2 + layer3 + max over K neighbors.
// WG = 256 thr (8 waves) handles 4 queries -> M = 96 rows (= 6 m-tiles).
// dynamic LDS 113152 B
// ---------------------------------------------------------------------------
__global__ __launch_bounds__(256) void fused_mlp_kernel(
    const float* __restrict__ p, const float* __restrict__ R,
    const float* __restrict__ W1, const float* __restrict__ b1,
    const float* __restrict__ b2, const float* __restrict__ b3,
    const float* __restrict__ U, const int* __restrict__ knn_idx,
    const bf16_t* __restrict__ Wt2, const bf16_t* __restrict__ Wt3,
    float* __restrict__ out) {
  extern __shared__ char smem[];
  bf16_t* xA    = (bf16_t*)smem;             // 96 x 264
  bf16_t* xB    = xA + 96 * 264;             // 96 x 264
  bf16_t* Btile = xB + 96 * 264;             // 2 x (64 x 40)
  float*  pij   = (float*)(Btile + 2 * 64 * 40);  // 96 x 3
  int*    jbuf  = (int*)(pij + 96 * 3);           // 96
  float*  stg   = (float*)smem;              // layer-3 staging, reuses xA

  const int t = threadIdx.x;
  const int wave = t >> 5, lane = t & 31, lm = lane & 15, hi = lane >> 4;
  const int ntl = wave & 3;
  const int mt0 = (wave >> 2) * 3;
  const int gq0 = blockIdx.x * 4;            // first global point row (b*N+n)

  // ---- phase 0: p_ij = R_i^T (p_j - p_i); prefetch neighbor T-rows ----
  if (t < 96) {
    const int q = t / KNN, kk = t % KNN;
    const int gi = gq0 + q;
    const int b = gi >> 12;
    const int jn = knn_idx[(size_t)gi * KNN + kk];
    const int gj = (b << 12) + jn;
    __builtin_prefetch((const void*)(U + (size_t)gj * 512 + 256), 0, 3);
    const float dx = p[gj * 3 + 0] - p[gi * 3 + 0];
    const float dy = p[gj * 3 + 1] - p[gi * 3 + 1];
    const float dz = p[gj * 3 + 2] - p[gi * 3 + 2];
    const float* Ri = R + (size_t)gi * 9;
    pij[t * 3 + 0] = Ri[0] * dx + Ri[3] * dy + Ri[6] * dz;
    pij[t * 3 + 1] = Ri[1] * dx + Ri[4] * dy + Ri[7] * dz;
    pij[t * 3 + 2] = Ri[2] * dx + Ri[5] * dy + Ri[8] * dz;
    jbuf[t] = gj;
  }
  __syncthreads();

  // ---- phase 1: x1 = leaky(S[i] + T[j] + p_ij.W1p + b1) -> xA (bf16) ----
  {
    const int c = t;  // one column per thread (coalesced U-row reads)
    const float w0 = W1[c], w1 = W1[256 + c], w2 = W1[512 + c];
    const float bb1 = b1[c];
    for (int q = 0; q < 4; ++q) {
      const float sreg = U[(size_t)(gq0 + q) * 512 + c];
#pragma unroll 4
      for (int kk = 0; kk < KNN; ++kk) {
        const int r = q * KNN + kk;
        const float tv = U[(size_t)jbuf[r] * 512 + 256 + c];
        float x = sreg + tv + bb1 + pij[r * 3 + 0] * w0 + pij[r * 3 + 1] * w1 +
                  pij[r * 3 + 2] * w2;
        x = fmaxf(x, 0.1f * x);
        xA[r * 264 + c] = (bf16_t)x;
      }
    }
  }
  __syncthreads();

  // ---- phase 2: x2 = leaky(x1 @ W2 + b2) -> xB (bf16) ----
  for (int ch = 0; ch < 4; ++ch) {
    v8f acc[3] = {};
    gemm_chunk<3>(xA, Wt2, Btile, ch * 64, mt0, ntl, lm, hi, t, acc);
    const int col = ch * 64 + ntl * 16 + lm;
    const float bb = b2[col];
#pragma unroll
    for (int mi = 0; mi < 3; ++mi)
#pragma unroll
      for (int r = 0; r < 8; ++r) {
        const int row = (mt0 + mi) * 16 + r + 8 * hi;
        float x = acc[mi][r] + bb;
        x = fmaxf(x, 0.1f * x);
        xB[row * 264 + col] = (bf16_t)x;
      }
  }
  __syncthreads();

  // ---- phase 3: x3 = x2 @ W3; out = max_k(x3) + b3 ----
  for (int ch = 0; ch < 4; ++ch) {
    v8f acc[3] = {};
    gemm_chunk<3>(xB, Wt3, Btile, ch * 64, mt0, ntl, lm, hi, t, acc);
    const int colrel = ntl * 16 + lm;
#pragma unroll
    for (int mi = 0; mi < 3; ++mi)
#pragma unroll
      for (int r = 0; r < 8; ++r)
        stg[((mt0 + mi) * 16 + r + 8 * hi) * 68 + colrel] = acc[mi][r];
    __syncthreads();
    {
      const int q = t >> 6, cc = t & 63;  // 256 outputs this chunk
      float m = -3.0e38f;
#pragma unroll 4
      for (int kk = 0; kk < KNN; ++kk)
        m = fmaxf(m, stg[(q * KNN + kk) * 68 + cc]);
      const int col = ch * 64 + cc;
      out[(size_t)(gq0 + q) * HD + col] = m + b3[col];
    }
    __syncthreads();
  }
}

// ---------------------------------------------------------------------------
// Host launcher
// ---------------------------------------------------------------------------
extern "C" void kernel_launch(void* const* d_in, const int* in_sizes, int n_in,
                              void* d_out, int out_size, void* d_ws, size_t ws_size,
                              hipStream_t stream) {
  const float* p  = (const float*)d_in[0];
  const float* R  = (const float*)d_in[1];
  const float* h  = (const float*)d_in[2];
  const float* W1 = (const float*)d_in[3];
  const float* b1 = (const float*)d_in[4];
  const float* W2 = (const float*)d_in[5];
  const float* b2 = (const float*)d_in[6];
  const float* W3 = (const float*)d_in[7];
  const float* b3 = (const float*)d_in[8];
  float* out = (float*)d_out;

  char* ws = (char*)d_ws;
  bf16_t* hbf = (bf16_t*)(ws + 0);          //  4,194,304 B
  bf16_t* Bt1 = (bf16_t*)(ws + 4194304);    //    262,144 B
  bf16_t* Bt2 = (bf16_t*)(ws + 4456448);    //    131,072 B
  bf16_t* Bt3 = (bf16_t*)(ws + 4587520);    //    131,072 B
  float*  sq  = (float*) (ws + 4718592);    //     32,768 B
  int*    idx = (int*)   (ws + 4751360);    //    786,432 B
  float*  U   = (float*) (ws + 5537792);    // 16,777,216 B  (total ~22.3 MB)

  pack_h_kernel<<<NROWS, 256, 0, stream>>>(h, hbf, sq);
  pack_w_kernel<<<1024, 256, 0, stream>>>(W1, W2, W3, Bt1, Bt2, Bt3);

  const size_t knn_lds =
      (64 * 260 + 2 * 32 * 260 + 64 + 4 * 16 * 33 + 64 * KNN) * 4 + 64 * KNN * 4;
  knn_kernel<<<NB * (NPTS / 64), 128, knn_lds, stream>>>(h, sq, idx);

  gemm_u_kernel<<<(NROWS / 128) * 4, 256, 0, stream>>>(hbf, Bt1, U);

  const size_t fused_lds =
      (size_t)(2 * 96 * 264 + 2 * 64 * 40) * 2 + 96 * 3 * 4 + 96 * 4;
  fused_mlp_kernel<<<NROWS / 4, 256, fused_lds, stream>>>(p, R, W1, b1, b2, b3, U, idx,
                                                          Bt2, Bt3, out);
}
